// MultiheadAttention_60945585930934
// MI455X (gfx1250) — matrix operand order repaired and checked by reading, not compiled, not run on hardware
//
#include <hip/hip_runtime.h>
#include <hip/hip_bf16.h>

typedef __bf16 bf16;
typedef __attribute__((ext_vector_type(16))) __bf16 v16bf;
typedef __attribute__((ext_vector_type(8)))  __bf16 v8bf;
typedef __attribute__((ext_vector_type(8)))  float  v8f;
typedef __attribute__((ext_vector_type(4)))  int    v4i;

#define WMMA_BF16(a, b, c) \
    __builtin_amdgcn_wmma_f32_16x16x32_bf16(false, (a), false, (b), (short)0, (c), false, false)

// ---------------------------------------------------------------------------
// gfx1250 async global->LDS copy (ASYNCcnt path), guarded with fallback.
// Builtin signature (from hipcc diagnostic): (v4i as1* gsrc, v4i as3* lds,
// imm offset, imm cpol).
// ---------------------------------------------------------------------------
#if __has_builtin(__builtin_amdgcn_global_load_async_to_lds_b128)
#define ASYNC_LDS 1
#else
#define ASYNC_LDS 0
#endif

__device__ __forceinline__ void cp_lds16(bf16* lds_dst, const bf16* gsrc) {
#if ASYNC_LDS
    __builtin_amdgcn_global_load_async_to_lds_b128(
        (v4i __attribute__((address_space(1)))*)gsrc,
        (v4i __attribute__((address_space(3)))*)lds_dst,
        0, 0);
#else
    *(v8bf*)lds_dst = *(const v8bf*)gsrc;
#endif
}

__device__ __forceinline__ void cp_lds_wait() {
#if ASYNC_LDS
#if __has_builtin(__builtin_amdgcn_s_wait_asynccnt)
    __builtin_amdgcn_s_wait_asynccnt(0);
#else
    asm volatile("s_wait_asynccnt 0" ::: "memory");
#endif
#endif
}

// ---------------------------------------------------------------------------
// Fragment loaders (contiguous k = 16*half + s in both A and B, 16-bit, wave32)
// ---------------------------------------------------------------------------

// A-matrix 16x32 (MxK), row-major source with leading dim `ld` (elements).
__device__ __forceinline__ v16bf load_a_frag(const bf16* tile, int ld, int k0, int lane) {
    const int row  = lane & 15;
    const int half = lane >> 4;
    return *(const v16bf*)(tile + (size_t)row * ld + k0 + half * 16);   // K = half*16 + 0..15
}

// B-matrix 32x16 (KxN) where B[k][n] = mat[n0+n][k0+k], mat row-major ld.
__device__ __forceinline__ v16bf load_b_frag(const bf16* mat, int ld, int n0, int k0, int lane) {
    const int col  = lane & 15;
    const int half = lane >> 4;
    return *(const v16bf*)(mat + (size_t)(n0 + col) * ld + k0 + half * 16);
}

// ---------------------------------------------------------------------------
// Kernel 0: f32 -> bf16 conversion
// ---------------------------------------------------------------------------
__global__ void cvt_f32_bf16_kernel(const float* __restrict__ src, bf16* __restrict__ dst, int n) {
    int i = blockIdx.x * blockDim.x + threadIdx.x;
    if (i < n) dst[i] = (bf16)src[i];
}

// ---------------------------------------------------------------------------
// Shared GEMM mainloop: C[128x128] block tile, 8 waves of 32x64, K-step 32,
// double-buffered LDS staging via async global->LDS copies.
// A: [M,K] row-major (ld=K).  B: [N,K] row-major (ld=K), used as B^T.
// ---------------------------------------------------------------------------
#define TILE_ELEMS (128 * 32)

__device__ __forceinline__ void gemm_block_128x128(
    const bf16* __restrict__ A, const bf16* __restrict__ Bm, int K,
    int m_block, int n_block, bf16* sA, bf16* sB, v8f acc[2][4])
{
    const int tid  = threadIdx.x;
    const int lane = tid & 31;
    const int wave = tid >> 5;
    const int wm   = wave & 3;          // 4 waves along M (32 rows each)
    const int wn   = wave >> 2;         // 2 waves along N (64 cols each)

    // staging: 256 threads x 16B = half of a 128x32 bf16 tile (k 0-7, 16-23 per row)
    const int srow = tid >> 1;          // 0..127
    const int skp  = (tid & 1) * 16;    // 0 or 16

    const bf16* gA = A  + (size_t)(m_block + srow) * K + skp;
    const bf16* gB = Bm + (size_t)(n_block + srow) * K + skp;
    const int soff = srow * 32 + skp;

    // prologue: stage tile k0=0 into buffer 0
    cp_lds16(sA + soff, gA);
    cp_lds16(sB + soff, gB);
    cp_lds_wait();
    __syncthreads();

    int p = 0;
    for (int k0 = 0; k0 < K; k0 += 32) {
        // prefetch next tile into the other buffer
        if (k0 + 32 < K) {
            cp_lds16(sA + (p ^ 1) * TILE_ELEMS + soff, gA + k0 + 32);
            cp_lds16(sB + (p ^ 1) * TILE_ELEMS + soff, gB + k0 + 32);
        }
        const bf16* tA = sA + p * TILE_ELEMS;
        const bf16* tB = sB + p * TILE_ELEMS;
        v16bf a0 = load_a_frag(tA + (wm * 32) * 32,      32, 0, lane);
        v16bf a1 = load_a_frag(tA + (wm * 32 + 16) * 32, 32, 0, lane);
#pragma unroll
        for (int j = 0; j < 4; ++j) {
            v16bf b = load_b_frag(tB, 32, wn * 64 + j * 16, 0, lane);
            acc[0][j] = WMMA_BF16(a0, b, acc[0][j]);
            acc[1][j] = WMMA_BF16(a1, b, acc[1][j]);
        }
        cp_lds_wait();
        __syncthreads();
        p ^= 1;
    }
}

// ---------------------------------------------------------------------------
// Kernel 1: qkv = x @ ipw.T + ipb, fused RoPE, scatter to q/k [B,H,T,64] and
// v transposed [B,H,64,T].  Grid: 64 m-blocks x 24 n-blocks.
// ---------------------------------------------------------------------------
__global__ __launch_bounds__(256)
void qkv_rope_kernel(const bf16* __restrict__ xb, const bf16* __restrict__ wb,
                     const float* __restrict__ bias,
                     bf16* __restrict__ qh, bf16* __restrict__ kh, bf16* __restrict__ vT)
{
    __shared__ __align__(16) bf16 sA[2 * TILE_ELEMS];
    __shared__ __align__(16) bf16 sB[2 * TILE_ELEMS];

    const int lane = threadIdx.x & 31;
    const int wave = threadIdx.x >> 5;
    const int wm   = wave & 3;
    const int wn   = wave >> 2;
    const int m_block = (blockIdx.x & 63) * 128;
    const int n_block = (blockIdx.x >> 6) * 128;

    v8f acc[2][4] = {};
    gemm_block_128x128(xb, wb, 1024, m_block, n_block, sA, sB, acc);

    const int half = lane >> 4;
    const int col  = lane & 15;
#pragma unroll
    for (int i = 0; i < 2; ++i) {
        const int mr0 = m_block + wm * 32 + i * 16;
#pragma unroll
        for (int j = 0; j < 4; ++j) {
            const int n_abs = n_block + wn * 64 + j * 16 + col;
            const int kind  = n_abs >> 10;     // 0=q 1=k 2=v (uniform per wave)
            const int nn    = n_abs & 1023;
            const int h     = nn >> 6;         // uniform per wave
            const int d     = nn & 63;
            const float bia = bias[n_abs];
#pragma unroll
            for (int e = 0; e < 8; ++e) {
                const int m = mr0 + e + half * 8;
                const int t = m & 1023;
                const int bb = m >> 10;
                float val = acc[i][j][e] + bia;
                if (kind < 2) {
                    // RoPE: pairs (2i,2i+1) are adjacent cols -> adjacent lanes
                    float partner = __shfl_xor(val, 1, 32);
                    const int i2 = d & ~1;
                    float inv = __expf(-9.210340371976184f * (float)i2 * (1.0f / 64.0f));
                    float ang = (float)t * inv;
                    float cs = __cosf(ang), sn = __sinf(ang);
                    float rot = (d & 1) ? partner : -partner;
                    val = val * cs + rot * sn;
                    bf16* dst = (kind == 0) ? qh : kh;
                    dst[((size_t)(bb * 16 + h) * 1024 + t) * 64 + d] = (bf16)val;
                } else {
                    vT[((size_t)(bb * 16 + h) * 64 + d) * 1024 + t] = (bf16)val;
                }
            }
        }
    }
}

// ---------------------------------------------------------------------------
// Kernel 2: flash attention per (b,h).  Block = 8 waves x 16 q-rows.
// ---------------------------------------------------------------------------
__global__ __launch_bounds__(256)
void attn_kernel(const bf16* __restrict__ qh, const bf16* __restrict__ kh,
                 const bf16* __restrict__ vT, bf16* __restrict__ yh)
{
    __shared__ __align__(16) bf16 pbuf[8][16 * 32];

    const int lane = threadIdx.x & 31;
    const int wave = threadIdx.x >> 5;
    const int qblk = blockIdx.x & 7;
    const int bh   = blockIdx.x >> 3;          // b*16 + h
    const int qr0  = qblk * 128 + wave * 16;

    const bf16* qp = qh + (size_t)bh * 1024 * 64;
    const bf16* kp = kh + (size_t)bh * 1024 * 64;
    const bf16* vp = vT + (size_t)bh * 64 * 1024;

    const int half = lane >> 4;
    const int col  = lane & 15;

    v8f yacc[4] = {};
    float mrow[8], lrow[8];
#pragma unroll
    for (int e = 0; e < 8; ++e) { mrow[e] = -3.0e38f; lrow[e] = 0.0f; }

    const v16bf aq0 = load_a_frag(qp + (size_t)qr0 * 64, 64, 0,  lane);
    const v16bf aq1 = load_a_frag(qp + (size_t)qr0 * 64, 64, 32, lane);

    const int kend = qr0 + 16;                 // causal bound
    for (int kb = 0; kb < kend; kb += 32) {
        v8f s0 = {}, s1 = {};
        {
            v16bf b00 = load_b_frag(kp, 64, kb,      0,  lane);
            v16bf b10 = load_b_frag(kp, 64, kb + 16, 0,  lane);
            s0 = WMMA_BF16(aq0, b00, s0);
            s1 = WMMA_BF16(aq0, b10, s1);
            v16bf b01 = load_b_frag(kp, 64, kb,      32, lane);
            v16bf b11 = load_b_frag(kp, 64, kb + 16, 32, lane);
            s0 = WMMA_BF16(aq1, b01, s0);
            s1 = WMMA_BF16(aq1, b11, s1);
        }

#pragma unroll
        for (int e = 0; e < 8; ++e) {
            const int r  = qr0 + e + half * 8;
            const int c0 = kb + col;
            const int c1 = kb + 16 + col;
            float v0 = s0[e] * 0.125f;
            float v1 = s1[e] * 0.125f;
            if (c0 > r) v0 = -1.0e9f;
            if (c1 > r) v1 = -1.0e9f;
            float mx = fmaxf(v0, v1);
#pragma unroll
            for (int o = 8; o > 0; o >>= 1) mx = fmaxf(mx, __shfl_xor(mx, o, 32));
            const float mnew = fmaxf(mrow[e], mx);
            const float corr = __expf(mrow[e] - mnew);
            const float p0 = __expf(v0 - mnew);
            const float p1 = __expf(v1 - mnew);
            float ps = p0 + p1;
#pragma unroll
            for (int o = 8; o > 0; o >>= 1) ps += __shfl_xor(ps, o, 32);
            lrow[e] = lrow[e] * corr + ps;
            mrow[e] = mnew;
#pragma unroll
            for (int j = 0; j < 4; ++j) yacc[j][e] *= corr;
            pbuf[wave][(e + half * 8) * 32 + col]      = (bf16)p0;
            pbuf[wave][(e + half * 8) * 32 + col + 16] = (bf16)p1;
        }

        asm volatile("s_wait_dscnt 0" ::: "memory");   // wave-private LDS handoff

        v16bf pa = load_a_frag(&pbuf[wave][0], 32, 0, lane);
#pragma unroll
        for (int j = 0; j < 4; ++j) {
            v16bf bv = load_b_frag(vp, 1024, j * 16, kb, lane);
            yacc[j] = WMMA_BF16(pa, bv, yacc[j]);
        }
    }

    const int bb = bh >> 4;
    const int h  = bh & 15;
#pragma unroll
    for (int j = 0; j < 4; ++j) {
#pragma unroll
        for (int e = 0; e < 8; ++e) {
            const int r = qr0 + e + half * 8;
            float y = yacc[j][e] / lrow[e];
            yh[((size_t)(bb * 1024 + r)) * 1024 + h * 64 + j * 16 + col] = (bf16)y;
        }
    }
}

// ---------------------------------------------------------------------------
// Kernel 3: out = y @ opw.T + opb.  Grid: 64 m-blocks x 8 n-blocks, f32 out.
// ---------------------------------------------------------------------------
__global__ __launch_bounds__(256)
void proj_kernel(const bf16* __restrict__ yb, const bf16* __restrict__ wb,
                 const float* __restrict__ bias, float* __restrict__ out)
{
    __shared__ __align__(16) bf16 sA[2 * TILE_ELEMS];
    __shared__ __align__(16) bf16 sB[2 * TILE_ELEMS];

    const int lane = threadIdx.x & 31;
    const int wave = threadIdx.x >> 5;
    const int wm   = wave & 3;
    const int wn   = wave >> 2;
    const int m_block = (blockIdx.x & 63) * 128;
    const int n_block = (blockIdx.x >> 6) * 128;

    v8f acc[2][4] = {};
    gemm_block_128x128(yb, wb, 1024, m_block, n_block, sA, sB, acc);

    const int half = lane >> 4;
    const int col  = lane & 15;
#pragma unroll
    for (int i = 0; i < 2; ++i) {
        const int mr0 = m_block + wm * 32 + i * 16;
#pragma unroll
        for (int j = 0; j < 4; ++j) {
            const int n = n_block + wn * 64 + j * 16 + col;
            const float bia = bias[n];
#pragma unroll
            for (int e = 0; e < 8; ++e) {
                const int m = mr0 + e + half * 8;
                out[(size_t)m * 1024 + n] = acc[i][j][e] + bia;
            }
        }
    }
}

// ---------------------------------------------------------------------------
// Host-side launcher
// ---------------------------------------------------------------------------
extern "C" void kernel_launch(void* const* d_in, const int* in_sizes, int n_in,
                              void* d_out, int out_size, void* d_ws, size_t ws_size,
                              hipStream_t stream) {
    const float* x   = (const float*)d_in[0];   // [8,1024,1024]
    const float* ipw = (const float*)d_in[1];   // [3072,1024]
    const float* ipb = (const float*)d_in[2];   // [3072]
    const float* opw = (const float*)d_in[3];   // [1024,1024]
    const float* opb = (const float*)d_in[4];   // [1024]
    // d_in[5] attn_mask: pure causal, computed inline

    const size_t M   = 8ull * 1024 * 1024;      // 8192 x 1024 elements
    bf16* xb   = (bf16*)d_ws;
    bf16* ipwb = xb   + M;
    bf16* opwb = ipwb + 3ull * 1024 * 1024;
    bf16* qhb  = opwb + 1ull * 1024 * 1024;     // [B,H,T,64]
    bf16* khb  = qhb  + M;
    bf16* vTb  = khb  + M;                      // [B,H,64,T]
    bf16* yhb  = vTb  + M;                      // [B*T, C]

    const int nx = 8 * 1024 * 1024, nip = 3072 * 1024, nop = 1024 * 1024;
    cvt_f32_bf16_kernel<<<(nx  + 255) / 256, 256, 0, stream>>>(x,   xb,   nx);
    cvt_f32_bf16_kernel<<<(nip + 255) / 256, 256, 0, stream>>>(ipw, ipwb, nip);
    cvt_f32_bf16_kernel<<<(nop + 255) / 256, 256, 0, stream>>>(opw, opwb, nop);

    // QKV: 64 m-blocks x 24 n-blocks
    qkv_rope_kernel<<<64 * 24, 256, 0, stream>>>(xb, ipwb, ipb, qhb, khb, vTb);

    // attention: 8 batches * 16 heads * 8 q-blocks
    attn_kernel<<<1024, 256, 0, stream>>>(qhb, khb, vTb, yhb);

    // out-proj: 64 m-blocks x 8 n-blocks
    proj_kernel<<<64 * 8, 256, 0, stream>>>(yhb, opwb, opb, (float*)d_out);
}
